// SelfAttention_74620761801191
// MI455X (gfx1250) — matrix operand order repaired and checked by reading, not compiled, not run on hardware
//
#include <hip/hip_runtime.h>

// ---------------------------------------------------------------------------
// Self-attention (B=1, T=4096, C=768, H=12, hd=64), fp32 in/out.
// Compute-bound (~71 GFLOP vs ~40MB traffic) -> bf16 WMMA everywhere
// (v_wmma_f32_16x16x32_bf16), f32 accumulation, flash-attention inner loop.
// LDS fragment loads are 16B-aligned ds_load_b128; attention Q/K tiles are
// fetched by the Tensor Data Mover with double-buffered K (TENSORcnt pipelining).
// ---------------------------------------------------------------------------

#define TQ   4096
#define CDIM 768
#define NH   12
#define HD   64

typedef __attribute__((ext_vector_type(16))) __bf16         v16bf;
typedef __attribute__((ext_vector_type(8)))  unsigned short v8us;
typedef __attribute__((ext_vector_type(4)))  unsigned short v4us;
typedef __attribute__((ext_vector_type(8)))  float          v8f;
typedef __attribute__((ext_vector_type(4)))  float          v4f;
typedef __attribute__((ext_vector_type(4)))  unsigned int   v4u;
typedef __attribute__((ext_vector_type(8)))  int            v8i;
typedef __attribute__((ext_vector_type(4)))  int            v4i;

#if defined(__has_builtin)
#if __has_builtin(__builtin_amdgcn_tensor_load_to_lds) && __has_builtin(__builtin_amdgcn_s_wait_tensorcnt)
#define HAVE_TDM 1
#endif
#endif
#ifndef HAVE_TDM
#define HAVE_TDM 0
#endif

// Native fp32 -> bf16 (lowers to v_cvt_pk_bf16_f32 / v_fma_mix*_bf16).
__device__ __forceinline__ unsigned short f2bfu(float f) {
    __bf16 b = (__bf16)f;
    union { __bf16 b; unsigned short u; } c; c.b = b;
    return c.u;
}

__device__ __forceinline__ v4us cvt4(float a, float b, float c, float d) {
    v4us r;
    r[0] = f2bfu(a); r[1] = f2bfu(b); r[2] = f2bfu(c); r[3] = f2bfu(d);
    return r;
}

// A fragment: 16x32 bf16 from row-major [m][k] (ld in ushorts, rows 16B-aligned).
// Per ISA: lane m holds K = (lane>>4)*8 + {0..7} and 16 + same -> two b128 loads.
__device__ __forceinline__ v16bf load_a_frag(const unsigned short* base, int ld, int lane) {
    const unsigned short* p = base + (lane & 15) * ld + ((lane >> 4) << 3);
    union { v8us h[2]; v16bf b; } u;
    u.h[0] = *(const v8us*)(p);
    u.h[1] = *(const v8us*)(p + 16);
    return u.b;
}

// B fragment: 32x16 bf16 where the source is stored K-contiguous, i.e. [n][k]
// (row-major K tiles work directly for Q.K^T; V is stored transposed).
// Per ISA: lane holds col n = lane&15, K = (lane>>4)*8 + {0..7} and 16 + same -> two b128 loads.
__device__ __forceinline__ v16bf load_bk_frag(const unsigned short* base, int ld, int lane) {
    const unsigned short* p = base + (lane & 15) * ld + ((lane >> 4) << 3);
    union { v8us h[2]; v16bf b; } u;
    u.h[0] = *(const v8us*)(p);
    u.h[1] = *(const v8us*)(p + 16);
    return u.b;
}

__device__ __forceinline__ v8f wmma_bf16(v16bf a, v16bf b, v8f c) {
    return __builtin_amdgcn_wmma_f32_16x16x32_bf16(false, a, false, b, (short)0, c, false, false);
}

#if HAVE_TDM
// TDM: copy `rows` x 64 bf16 tile (row stride 64 elems in memory) into LDS with
// +8-ushort row padding (pad_interval: 32 DW -> code 4, pad_amount: 4 DW -> code 3).
__device__ __forceinline__ void tdm_load_tile(const unsigned short* gsrc,
                                              unsigned lds_byte_off, int rows) {
    unsigned long long ga = (unsigned long long)(uintptr_t)gsrc;
    v4u g0 = { 1u,                                   // count=1 (valid descriptor)
               lds_byte_off,                         // lds_addr
               (unsigned)ga,                         // global_addr[31:0]
               ((unsigned)(ga >> 32) & 0x01FFFFFFu) | 0x80000000u }; // [56:32] | type=2
    v8i g1 = { (int)((1u << 16) | (1u << 20) | (4u << 22) | (3u << 25)), // 2B elems, pad
               (int)(64u << 16),                     // tensor_dim0 = 64
               (int)(((unsigned)rows & 0xFFFFu) << 16), // tensor_dim1 = rows
               (int)(64u << 16),                     // tile_dim0 = 64
               rows,                                 // tile_dim1
               64, 0, 0 };                           // tensor_dim0_stride = 64
    v4i g2 = {0, 0, 0, 0}, g3 = {0, 0, 0, 0};
    v8i g4 = {0, 0, 0, 0, 0, 0, 0, 0};
    __builtin_amdgcn_tensor_load_to_lds(g0, g1, g2, g3, g4, 0);
}
#endif

// ---------------------------------------------------------------------------
// Kernel 1: qkv = x @ w_attn  [4096x768] x [768x2304] -> bf16 [mat][head][T][64]
// (q pre-scaled by 1/sqrt(64)).  Block tile 64x64, 8 waves x (16x32), K-step 64.
// ---------------------------------------------------------------------------
__global__ __launch_bounds__(256) void qkv_gemm(const float* __restrict__ x,
                                                const float* __restrict__ w,
                                                unsigned short* __restrict__ qkv) {
    const int LD = 72;
    __shared__ __align__(16) unsigned short As[64 * 72];   // [m][k]
    __shared__ __align__(16) unsigned short Bs[64 * 72];   // [n][k] (transposed)
    const int tid = threadIdx.x, lane = tid & 31, wave = tid >> 5;
    const int bm = blockIdx.y * 64, bn = blockIdx.x * 64;
    const int mrow = (wave & 3) * 16, ncol = (wave >> 2) * 32;

    // Per-thread staging coordinates (counted loops -> no exec guards).
    const int ar  = tid >> 4, ac4 = (tid & 15) << 2;   // A: 16 rows apart per j
    const int bc  = tid & 63, br4 = (tid >> 6) << 2;   // B: 16 k-rows apart per j

    v8f acc0 = {}, acc1 = {};
    for (int k0 = 0; k0 < CDIM; k0 += 64) {
        // Batch all global loads first, then convert + store (loads overlap).
        v4f av[4];
        float bv[4][4];
#pragma unroll
        for (int j = 0; j < 4; ++j)
            av[j] = *(const v4f*)(x + (bm + ar + j * 16) * CDIM + k0 + ac4);
#pragma unroll
        for (int j = 0; j < 4; ++j) {
            const float* wp = w + (k0 + br4 + j * 16) * (3 * CDIM) + bn + bc;
#pragma unroll
            for (int e = 0; e < 4; ++e) bv[j][e] = wp[e * 3 * CDIM];
        }
#pragma unroll
        for (int j = 0; j < 4; ++j)
            *(v4us*)(As + (ar + j * 16) * LD + ac4) =
                cvt4(av[j][0], av[j][1], av[j][2], av[j][3]);
#pragma unroll
        for (int j = 0; j < 4; ++j)
            *(v4us*)(Bs + bc * LD + br4 + j * 16) =
                cvt4(bv[j][0], bv[j][1], bv[j][2], bv[j][3]);
        __syncthreads();
#pragma unroll
        for (int ks = 0; ks < 64; ks += 32) {
            v16bf a  = load_a_frag(As + mrow * LD + ks, LD, lane);
            v16bf b0 = load_bk_frag(Bs + ncol * LD + ks, LD, lane);
            v16bf b1 = load_bk_frag(Bs + (ncol + 16) * LD + ks, LD, lane);
            acc0 = wmma_bf16(a, b0, acc0);
            acc1 = wmma_bf16(a, b1, acc1);
        }
        __syncthreads();
    }

    // Block-uniform output slot: bn is 64-aligned and one block = one (mat,head).
    const int mat = bn / CDIM;               // scalar
    const int hh  = (bn % CDIM) >> 6;        // scalar
    const float sc = (mat == 0) ? 0.125f : 1.0f;   // 1/sqrt(64) folded into q
    unsigned short* dstbase = qkv + (mat * NH + hh) * (TQ * HD);
    const int half = lane >> 4, n0 = lane & 15;
#pragma unroll
    for (int accn = 0; accn < 2; ++accn) {
        v8f A = accn ? acc1 : acc0;
        int d = ncol + accn * 16 + n0;
#pragma unroll
        for (int i = 0; i < 8; ++i) {
            int row = bm + mrow + i + 8 * half;
            dstbase[row * HD + d] = f2bfu(A[i] * sc);
        }
    }
}

// ---------------------------------------------------------------------------
// Kernel 2: flash attention; one head per blockIdx.y, 128 query rows per block,
// 8 waves each owning 16 rows (4 x 16x16 f32 accumulators over hd=64).
// Q + double-buffered K tiles via Tensor Data Mover; V transposed manually.
// ---------------------------------------------------------------------------
__global__ __launch_bounds__(256) void attn(const unsigned short* __restrict__ qkv,
                                            unsigned short* __restrict__ y) {
    const int LQ = 72;
    __shared__ __align__(16) unsigned short Qs[128 * 72];      // [m][d]
    __shared__ __align__(16) unsigned short Ks[2][64 * 72];    // [key][d], ping-pong
    __shared__ __align__(16) unsigned short Vs[64 * 72];       // [d][key] (transposed)
    __shared__ __align__(16) unsigned short Ps[8][16 * 72];    // per-wave P, [m][key]

    const int tid = threadIdx.x, lane = tid & 31, wave = tid >> 5;
    const int h = blockIdx.y;
    const int q0 = blockIdx.x * 128;
    const unsigned short* qb = qkv + (0 * NH + h) * (TQ * HD);
    const unsigned short* kb = qkv + (1 * NH + h) * (TQ * HD);
    const unsigned short* vb = qkv + (2 * NH + h) * (TQ * HD);

    const int rowTop = q0 + wave * 16 + 15;
    const int nkt = q0 / 64 + 2;

#if HAVE_TDM
    if (wave == 0) {
        tdm_load_tile(qb + q0 * HD, (unsigned)(uintptr_t)(void*)Qs, 128);
        tdm_load_tile(kb, (unsigned)(uintptr_t)(void*)Ks[0], 64);
        __builtin_amdgcn_s_wait_tensorcnt(1);   // in-order: Q has landed
    }
#else
#pragma unroll
    for (int j = 0; j < 32; ++j) {
        int i = tid + j * 256;
        int r = i >> 6, c = i & 63;
        Qs[r * LQ + c] = qb[(q0 + r) * HD + c];
    }
#endif
    __syncthreads();

    const int half = lane >> 4, n0 = lane & 15;
    float mrow[8], lrow[8];
    v8f O[4];
    v8f zero = {};
#pragma unroll
    for (int i = 0; i < 8; ++i) { mrow[i] = -1e30f; lrow[i] = 0.0f; }
#pragma unroll
    for (int c = 0; c < 4; ++c) O[c] = zero;

    const int vc = tid & 63, vr4 = (tid >> 6) << 2;   // V staging coords

    for (int kt = 0; kt < nkt; ++kt) {
        const unsigned short* Kcur = Ks[kt & 1];
        // V stored transposed [d][key]: batch 16 coalesced loads, then 4 b64 stores.
        unsigned short vv[4][4];
#pragma unroll
        for (int j = 0; j < 4; ++j) {
            const unsigned short* vp = vb + (kt * 64 + vr4 + j * 16) * HD + vc;
#pragma unroll
            for (int e = 0; e < 4; ++e) vv[j][e] = vp[e * HD];
        }
#pragma unroll
        for (int j = 0; j < 4; ++j) {
            v4us t = { vv[j][0], vv[j][1], vv[j][2], vv[j][3] };
            *(v4us*)(Vs + vc * LQ + vr4 + j * 16) = t;
        }
#if HAVE_TDM
        if (wave == 0) {
            if (kt + 1 < nkt) {   // prefetch next K tile into the other buffer
                tdm_load_tile(kb + (kt + 1) * 64 * HD,
                              (unsigned)(uintptr_t)(void*)Ks[(kt + 1) & 1], 64);
                __builtin_amdgcn_s_wait_tensorcnt(1);   // current tile complete
            } else {
                __builtin_amdgcn_s_wait_tensorcnt(0);
            }
        }
#else
#pragma unroll
        for (int j = 0; j < 16; ++j) {
            int i = tid + j * 256;
            int r = i >> 6, c = i & 63;
            Ks[kt & 1][r * LQ + c] = kb[(kt * 64 + r) * HD + c];
        }
#endif
        __syncthreads();

        bool active = (kt * 64) <= rowTop;   // skip fully-masked tiles
        if (active) {
            // S = Q . K^T  (16x64 per wave); K tile is [n=key][k=d] as needed.
            v8f S[4];
#pragma unroll
            for (int c = 0; c < 4; ++c) S[c] = zero;
#pragma unroll
            for (int ks = 0; ks < 2; ++ks) {
                int k0 = ks * 32;
                v16bf aq = load_a_frag(Qs + (wave * 16) * LQ + k0, LQ, lane);
#pragma unroll
                for (int c = 0; c < 4; ++c) {
                    v16bf bk = load_bk_frag(Kcur + (c * 16) * LQ + k0, LQ, lane);
                    S[c] = wmma_bf16(aq, bk, S[c]);
                }
            }
            // causal mask + online softmax (a row spans the 16 lanes of a half)
#pragma unroll
            for (int i = 0; i < 8; ++i) {
                int row = q0 + wave * 16 + i + 8 * half;
                float rmax = -1e30f;
#pragma unroll
                for (int c = 0; c < 4; ++c) {
                    int key = kt * 64 + c * 16 + n0;
                    float s = (key <= row) ? S[c][i] : -1e30f;
                    S[c][i] = s;
                    rmax = fmaxf(rmax, s);
                }
#pragma unroll
                for (int off = 8; off > 0; off >>= 1)
                    rmax = fmaxf(rmax, __shfl_xor(rmax, off, 16));
                float mn = fmaxf(mrow[i], rmax);
                float alpha = __expf(mrow[i] - mn);
                float rsum = 0.0f;
#pragma unroll
                for (int c = 0; c < 4; ++c) {
                    float p = __expf(S[c][i] - mn);
                    rsum += p;
                    Ps[wave][(i + 8 * half) * LQ + c * 16 + n0] = f2bfu(p);
                }
#pragma unroll
                for (int off = 8; off > 0; off >>= 1)
                    rsum += __shfl_xor(rsum, off, 16);
                mrow[i] = mn;
                lrow[i] = lrow[i] * alpha + rsum;
#pragma unroll
                for (int c = 0; c < 4; ++c) O[c][i] *= alpha;
            }
            // O += P . V  (wave-private Ps; compiler inserts DS waits)
#pragma unroll
            for (int ks = 0; ks < 2; ++ks) {
                int k0 = ks * 32;
                v16bf ap = load_a_frag(&Ps[wave][k0], LQ, lane);
#pragma unroll
                for (int c = 0; c < 4; ++c) {
                    v16bf bv2 = load_bk_frag(Vs + (c * 16) * LQ + k0, LQ, lane);
                    O[c] = wmma_bf16(ap, bv2, O[c]);
                }
            }
        }
        __syncthreads();
    }

    // normalize and write y as bf16 [T][C]
#pragma unroll
    for (int c = 0; c < 4; ++c) {
#pragma unroll
        for (int i = 0; i < 8; ++i) {
            int row = q0 + wave * 16 + i + 8 * half;
            int col = h * HD + c * 16 + n0;
            y[row * CDIM + col] = f2bfu(O[c][i] / lrow[i]);
        }
    }
}

// ---------------------------------------------------------------------------
// Kernel 3: out = y @ w_proj  [4096x768] x [768x768] -> fp32
// ---------------------------------------------------------------------------
__global__ __launch_bounds__(256) void proj_gemm(const unsigned short* __restrict__ y,
                                                 const float* __restrict__ w,
                                                 float* __restrict__ out) {
    const int LD = 72;
    __shared__ __align__(16) unsigned short As[64 * 72];   // [m][k]
    __shared__ __align__(16) unsigned short Bs[64 * 72];   // [n][k]
    const int tid = threadIdx.x, lane = tid & 31, wave = tid >> 5;
    const int bm = blockIdx.y * 64, bn = blockIdx.x * 64;
    const int mrow = (wave & 3) * 16, ncol = (wave >> 2) * 32;

    const int ar  = tid >> 3, ac8 = (tid & 7) << 3;    // A: b128 copies
    const int bc  = tid & 63, br4 = (tid >> 6) << 2;   // B: transposed store

    v8f acc0 = {}, acc1 = {};
    for (int k0 = 0; k0 < CDIM; k0 += 64) {
        v8us ac[2];
        float bv[4][4];
#pragma unroll
        for (int j = 0; j < 2; ++j)
            ac[j] = *(const v8us*)(y + (bm + ar + j * 32) * CDIM + k0 + ac8);
#pragma unroll
        for (int j = 0; j < 4; ++j) {
            const float* wp = w + (k0 + br4 + j * 16) * CDIM + bn + bc;
#pragma unroll
            for (int e = 0; e < 4; ++e) bv[j][e] = wp[e * CDIM];
        }
#pragma unroll
        for (int j = 0; j < 2; ++j)
            *(v8us*)(As + (ar + j * 32) * LD + ac8) = ac[j];
#pragma unroll
        for (int j = 0; j < 4; ++j)
            *(v4us*)(Bs + bc * LD + br4 + j * 16) =
                cvt4(bv[j][0], bv[j][1], bv[j][2], bv[j][3]);
        __syncthreads();
#pragma unroll
        for (int ks = 0; ks < 64; ks += 32) {
            v16bf a  = load_a_frag(As + mrow * LD + ks, LD, lane);
            v16bf b0 = load_bk_frag(Bs + ncol * LD + ks, LD, lane);
            v16bf b1 = load_bk_frag(Bs + (ncol + 16) * LD + ks, LD, lane);
            acc0 = wmma_bf16(a, b0, acc0);
            acc1 = wmma_bf16(a, b1, acc1);
        }
        __syncthreads();
    }

    const int half = lane >> 4, n0 = lane & 15;
#pragma unroll
    for (int accn = 0; accn < 2; ++accn) {
        v8f A = accn ? acc1 : acc0;
        int col = bn + ncol + accn * 16 + n0;
#pragma unroll
        for (int i = 0; i < 8; ++i) {
            int row = bm + mrow + i + 8 * half;
            out[row * CDIM + col] = A[i];
        }
    }
}

extern "C" void kernel_launch(void* const* d_in, const int* in_sizes, int n_in,
                              void* d_out, int out_size, void* d_ws, size_t ws_size,
                              hipStream_t stream) {
    const float* x      = (const float*)d_in[0];
    const float* w_attn = (const float*)d_in[1];
    const float* w_proj = (const float*)d_in[2];
    float* out = (float*)d_out;

    unsigned short* qkv = (unsigned short*)d_ws;                 // 3*12*4096*64 bf16
    unsigned short* yb  = qkv + 3 * NH * TQ * HD;                // 4096*768 bf16

    qkv_gemm<<<dim3(3 * CDIM / 64, TQ / 64), 256, 0, stream>>>(x, w_attn, qkv);
    attn<<<dim3(TQ / 128, NH), 256, 0, stream>>>(qkv, yb);
    proj_gemm<<<dim3(CDIM / 64, TQ / 64), 256, 0, stream>>>(yb, w_proj, out);
}
